// DTransformer_41412074668313
// MI455X (gfx1250) — matrix operand order repaired and checked by reading, not compiled, not run on hardware
//
#include <hip/hip_runtime.h>
#include <hip/hip_bf16.h>
#include <math.h>

// ---------------- problem constants (match reference) ----------------
#define DM   128      // d_model
#define HH   8        // heads
#define DKK  16       // head dim
#define NKN  16       // n_know
#define BSB  4        // batch
#define SS   256      // seq len
#define NEGV (-1e32f)

typedef __attribute__((ext_vector_type(16))) _Float16 v16h;
typedef __attribute__((ext_vector_type(8)))  float    v8f;

// =====================================================================
// WMMA tile loaders — layouts per CDNA5 ISA §7.12.2 (wave32)
//  A 16x32 f16 : lane L (row m=L&15): halves 0..7  = K kb..kb+7,
//                halves 8..15 = K kb+16..kb+23, kb = (L<16)?0:8
//  B 32x16 f16 : lane L (col n=L&15): halves 0..7  = K kb..kb+7,
//                halves 8..15 = K kb+16..kb+23, kb = (L<16)?0:8
//  C/D 16x16 f32: lane L (col n=L&15): vgpr r = row r + ((L<16)?0:8)
// =====================================================================

__device__ __forceinline__ v16h load_a_f32(const float* __restrict__ A, int lda,
                                           int k0, int kvalid) {
  int lane = threadIdx.x & 31;
  int m  = lane & 15;
  int kb = (lane < 16) ? 0 : 8;
  v16h a;
#pragma unroll
  for (int i = 0; i < 8; ++i) {
    int kr0 = kb + i;
    int kr1 = kb + 16 + i;
    a[i]     = (kr0 < kvalid) ? (_Float16)A[(size_t)m * lda + k0 + kr0] : (_Float16)0.f;
    a[8 + i] = (kr1 < kvalid) ? (_Float16)A[(size_t)m * lda + k0 + kr1] : (_Float16)0.f;
  }
  return a;
}

__device__ __forceinline__ v16h load_a_h16(const _Float16* __restrict__ A, int lda, int k0) {
  int lane = threadIdx.x & 31;
  int m  = lane & 15;
  int kb = (lane < 16) ? 0 : 8;
  v16h a;
#pragma unroll
  for (int i = 0; i < 8; ++i) {
    a[i]     = A[m * lda + k0 + kb + i];
    a[8 + i] = A[m * lda + k0 + kb + 16 + i];
  }
  return a;
}

// B[k][n] = W[n][k]  (transposed weight access: contiguous per lane)
__device__ __forceinline__ v16h load_bT_f32(const float* __restrict__ W, int ldw,
                                            int k0, int kvalid) {
  int lane = threadIdx.x & 31;
  int n  = lane & 15;
  int kb = (lane < 16) ? 0 : 8;
  v16h b;
#pragma unroll
  for (int i = 0; i < 16; ++i) {
    int kr = kb + i + ((i >= 8) ? 8 : 0);
    b[i] = (kr < kvalid) ? (_Float16)W[(size_t)n * ldw + k0 + kr] : (_Float16)0.f;
  }
  return b;
}

// B[k][n] = V[k][n]  (natural row-major V, strided per lane)
__device__ __forceinline__ v16h load_b_f32(const float* __restrict__ V, int ldv, int k0) {
  int lane = threadIdx.x & 31;
  int n  = lane & 15;
  int kb = (lane < 16) ? 0 : 8;
  v16h b;
#pragma unroll
  for (int i = 0; i < 16; ++i)
    b[i] = (_Float16)V[(size_t)(k0 + kb + i + ((i >= 8) ? 8 : 0)) * ldv + n];
  return b;
}

__device__ __forceinline__ v8f wmma_f16(v16h a, v16h b, v8f c) {
  return __builtin_amdgcn_wmma_f32_16x16x32_f16(false, a, false, b, (short)0, c,
                                                false, false);
}

// =====================================================================
// Kernel 1: Y(M x 128) = X(M x 128) @ W^T + bias   (grid: (M/16, 8), 32 thr)
// =====================================================================
__global__ void proj128_kernel(const float* __restrict__ X, const float* __restrict__ W,
                               const float* __restrict__ bias, float* __restrict__ Y) {
  int mt = blockIdx.x, nt = blockIdx.y;
  const float* Xt = X + (size_t)mt * 16 * DM;
  const float* Wt = W + (size_t)nt * 16 * DM;
  v8f acc = {};
#pragma unroll
  for (int k0 = 0; k0 < DM; k0 += 32) {
    __builtin_prefetch(Wt + ((threadIdx.x & 15) * DM) + k0, 0, 1);
    v16h a = load_a_f32(Xt, DM, k0, 32);
    v16h b = load_bT_f32(Wt, DM, k0, 32);
    acc = wmma_f16(a, b, acc);
  }
  int lane = threadIdx.x & 31;
  int n  = lane & 15;
  int mb = (lane < 16) ? 0 : 8;
  float bv = bias[nt * 16 + n];
#pragma unroll
  for (int r = 0; r < 8; ++r)
    Y[(size_t)(mt * 16 + r + mb) * DM + nt * 16 + n] = acc[r] + bv;
}

// =====================================================================
// per-row attention pipeline helpers (2 lanes per row: halves of 128 cols)
// =====================================================================
__device__ __forceinline__ float half_combine_max(float v) {
  return fmaxf(v, __shfl_xor(v, 16, 32));
}
__device__ __forceinline__ float half_other(float v) { return __shfl_xor(v, 16, 32); }

// =====================================================================
// Kernel 2: block-1 attention.  grid (S/16, H, BS), 32 threads.
// Q==K (kq_same, same input). mask: j<=i. maxout off.
// =====================================================================
__global__ void attn1_kernel(const float* __restrict__ Q1, const float* __restrict__ V1,
                             const float* __restrict__ gamma,
                             float* __restrict__ q_scores, float* __restrict__ attn_out) {
  __shared__ float    sc[16 * SS];
  __shared__ float    pr[16 * SS];
  __shared__ _Float16 ph[16 * SS];

  int it = blockIdx.x, h = blockIdx.y, b = blockIdx.z;
  int lane = threadIdx.x & 31;
  const float* Qh = Q1 + (size_t)b * SS * DM + h * DKK;
  const float* Vh = V1 + (size_t)b * SS * DM + h * DKK;
  float gam = fabsf(gamma[h]);

  // ---- raw scores (16 x 256) via WMMA, K=16 zero-padded to 32 ----
  v16h aq = load_a_f32(Qh + (size_t)it * 16 * DM, DM, 0, DKK);
  int n  = lane & 15;
  int mb = (lane < 16) ? 0 : 8;
  for (int jt = 0; jt < SS / 16; ++jt) {
    v16h bk = load_bT_f32(Qh + (size_t)jt * 16 * DM, DM, 0, DKK);
    v8f s = {};
    s = wmma_f16(aq, bk, s);
#pragma unroll
    for (int r = 0; r < 8; ++r) {
      int m  = r + mb;
      int ig = it * 16 + m;
      int jg = jt * 16 + n;
      sc[m * SS + jg] = (jg <= ig) ? s[r] * 0.25f : NEGV;
    }
  }
  __syncthreads();

  // ---- per-row softmax + time-decay + second softmax ----
  int r    = lane & 15;
  int half = lane >> 4;
  int j0   = half * 128;
  int ig   = it * 16 + r;
  float* row  = sc + r * SS;
  float* erow = pr + r * SS;

  float mx = NEGV;
  for (int j = j0; j < j0 + 128; ++j) mx = fmaxf(mx, row[j]);
  mx = half_combine_max(mx);

  float sm = 0.f;
  for (int j = j0; j < j0 + 128; ++j) { float e = expf(row[j] - mx); erow[j] = e; sm += e; }
  float smo   = half_other(sm);
  float inv   = 1.f / (sm + smo);
  float run   = half ? smo * inv : 0.f;   // exclusive prefix offset (disttotal == 1)

  float mx2 = NEGV;
  for (int j = j0; j < j0 + 128; ++j) {
    run += erow[j] * inv;                                   // cumsum of s_
    float pos  = fabsf((float)(j - ig));
    float dist = sqrtf(fmaxf((1.f - run) * pos, 0.f));
    float eff  = fminf(fmaxf(expf(-gam * dist), 1e-5f), 1e5f);
    float ns   = (j <= ig) ? row[j] * eff : NEGV;
    row[j] = ns;
    mx2 = fmaxf(mx2, ns);
  }
  mx2 = half_combine_max(mx2);

  float sm2 = 0.f;
  for (int j = j0; j < j0 + 128; ++j) { float e = expf(row[j] - mx2); erow[j] = e; sm2 += e; }
  float inv2 = 1.f / (sm2 + half_other(sm2));

  float* qs = q_scores + (((size_t)(b * HH + h) * SS + ig) * SS);
  for (int j = j0; j < j0 + 128; ++j) {
    float pn = (j <= ig) ? erow[j] * inv2 : 0.f;
    qs[j] = pn;
    ph[r * SS + j] = (_Float16)pn;
  }
  __syncthreads();

  // ---- out = P(16x256) @ Vh(256x16) via WMMA ----
  v8f acc = {};
#pragma unroll
  for (int k0 = 0; k0 < SS; k0 += 32) {
    v16h ap = load_a_h16(ph, SS, k0);
    v16h bv = load_b_f32(Vh, DM, k0);
    acc = wmma_f16(ap, bv, acc);
  }
#pragma unroll
  for (int rr = 0; rr < 8; ++rr) {
    int m = rr + mb;
    attn_out[((size_t)b * SS + it * 16 + m) * DM + h * DKK + n] = acc[rr];
  }
}

// =====================================================================
// Kernel 3: block-4 raw scores, exploiting broadcast structure:
//   cbase[b,h,nk,j] = dot(Qk[nk,h,:], K4[b,j,h,:]) / 4
//   grid (S/16, H, BS), 32 threads; M = NK = 16 exactly.
// =====================================================================
__global__ void score4_base_kernel(const float* __restrict__ Qk, const float* __restrict__ K4,
                                   float* __restrict__ cbase) {
  int jt = blockIdx.x, h = blockIdx.y, b = blockIdx.z;
  v16h a  = load_a_f32(Qk + h * DKK, DM, 0, DKK);
  v16h bb = load_bT_f32(K4 + ((size_t)b * SS + jt * 16) * DM + h * DKK, DM, 0, DKK);
  v8f s = {};
  s = wmma_f16(a, bb, s);
  int lane = threadIdx.x & 31;
  int n  = lane & 15;
  int mb = (lane < 16) ? 0 : 8;
#pragma unroll
  for (int r = 0; r < 8; ++r) {
    int nk = r + mb;
    cbase[((size_t)(b * HH + h) * NKN + nk) * SS + jt * 16 + n] = s[r] * 0.25f;
  }
}

// =====================================================================
// Kernel 4: block-4 attention. grid (S/16, H, BS*NK), 32 threads.
// Raw score row identical for every i (read from cbase); mask j<i; maxout on.
// k_scores layout: (b, h, i, nk, j)
// =====================================================================
__global__ void attn4_kernel(const float* __restrict__ cbase, const float* __restrict__ V4,
                             const float* __restrict__ gamma,
                             float* __restrict__ k_scores, float* __restrict__ attn_out) {
  __shared__ float    sc[16 * SS];
  __shared__ float    pr[16 * SS];
  __shared__ _Float16 ph[16 * SS];

  int it = blockIdx.x, h = blockIdx.y;
  int bn = blockIdx.z;
  int b  = bn / NKN, nk = bn % NKN;
  int lane = threadIdx.x & 31;
  float gam = fabsf(gamma[h]);

  const float* crow = cbase + ((size_t)(b * HH + h) * NKN + nk) * SS;
  const float* Vh   = V4 + (size_t)b * SS * DM + h * DKK;

  int r    = lane & 15;
  int half = lane >> 4;
  int j0   = half * 128;
  int ig   = it * 16 + r;
  float* row  = sc + r * SS;
  float* erow = pr + r * SS;

  // fill masked raw scores (strict causal: j < i)
  for (int j = j0; j < j0 + 128; ++j)
    row[j] = (j < ig) ? crow[j] : NEGV;

  float mx = NEGV;
  for (int j = j0; j < j0 + 128; ++j) mx = fmaxf(mx, row[j]);
  mx = half_combine_max(mx);

  float sm = 0.f;
  for (int j = j0; j < j0 + 128; ++j) { float e = expf(row[j] - mx); erow[j] = e; sm += e; }
  float smo = half_other(sm);
  float inv = 1.f / (sm + smo);
  float run = half ? smo * inv : 0.f;

  float mx2 = NEGV;
  for (int j = j0; j < j0 + 128; ++j) {
    run += erow[j] * inv;
    float pos  = fabsf((float)(j - ig));
    float dist = sqrtf(fmaxf((1.f - run) * pos, 0.f));
    float eff  = fminf(fmaxf(expf(-gam * dist), 1e-5f), 1e5f);
    float ns   = (j < ig) ? row[j] * eff : NEGV;
    row[j] = ns;
    mx2 = fmaxf(mx2, ns);
  }
  mx2 = half_combine_max(mx2);

  float sm2 = 0.f;
  for (int j = j0; j < j0 + 128; ++j) { float e = expf(row[j] - mx2); erow[j] = e; sm2 += e; }
  float inv2 = 1.f / (sm2 + half_other(sm2));

  // probabilities + maxout rescale
  float pmax = 0.f;
  for (int j = j0; j < j0 + 128; ++j) {
    float pn = (j < ig) ? erow[j] * inv2 : 0.f;
    erow[j] = pn;
    pmax = fmaxf(pmax, pn);
  }
  pmax = half_combine_max(pmax);
  float scale = fminf(1.f / pmax, 5.f);   // pmax==0 -> inf -> clamp 5, row stays 0

  float* ks = k_scores + ((((size_t)(b * HH + h) * SS + ig) * NKN + nk) * SS);
  for (int j = j0; j < j0 + 128; ++j) {
    float pv = erow[j] * scale;
    ks[j] = pv;
    ph[r * SS + j] = (_Float16)pv;
  }
  __syncthreads();

  // out = P @ V4h
  int n  = lane & 15;
  int mb = (lane < 16) ? 0 : 8;
  v8f acc = {};
#pragma unroll
  for (int k0 = 0; k0 < SS; k0 += 32) {
    v16h ap = load_a_h16(ph, SS, k0);
    v16h bv = load_b_f32(Vh, DM, k0);
    acc = wmma_f16(ap, bv, acc);
  }
#pragma unroll
  for (int rr = 0; rr < 8; ++rr) {
    int m = rr + mb;
    attn_out[((size_t)bn * SS + it * 16 + m) * DM + h * DKK + n] = acc[rr];
  }
}

// =====================================================================
// residual + layernorm (one wave per row of 128)
// =====================================================================
__device__ __forceinline__ void ln_row(const float* __restrict__ resid,
                                       const float* __restrict__ proj,
                                       const float* __restrict__ g,
                                       const float* __restrict__ be,
                                       float* __restrict__ out) {
  int lane = threadIdx.x & 31;
  float x[4];
  float s = 0.f;
#pragma unroll
  for (int i = 0; i < 4; ++i) { int d = lane * 4 + i; x[i] = resid[d] + proj[d]; s += x[i]; }
#pragma unroll
  for (int off = 16; off > 0; off >>= 1) s += __shfl_xor(s, off, 32);
  float mean = s * (1.f / (float)DM);
  float v = 0.f;
#pragma unroll
  for (int i = 0; i < 4; ++i) { float t = x[i] - mean; v += t * t; }
#pragma unroll
  for (int off = 16; off > 0; off >>= 1) v += __shfl_xor(v, off, 32);
  float rstd = rsqrtf(v * (1.f / (float)DM) + 1e-5f);
#pragma unroll
  for (int i = 0; i < 4; ++i) {
    int d = lane * 4 + i;
    out[d] = (x[i] - mean) * rstd * g[d] + be[d];
  }
}

__global__ void addln1_kernel(const float* __restrict__ resid, const float* __restrict__ proj,
                              const float* __restrict__ g, const float* __restrict__ be,
                              float* __restrict__ out) {
  size_t row = blockIdx.x;
  ln_row(resid + row * DM, proj + row * DM, g, be, out + row * DM);
}

// block-4: residual is know_params[nk]; output scattered to z layout (b,s,nk*D+d)
__global__ void addln4_kernel(const float* __restrict__ know, const float* __restrict__ proj,
                              const float* __restrict__ g, const float* __restrict__ be,
                              float* __restrict__ z) {
  size_t row = blockIdx.x;                 // row = (b*NK + nk)*S + s
  int s_  = (int)(row % SS);
  int nk  = (int)((row / SS) % NKN);
  int b   = (int)(row / ((size_t)NKN * SS));
  float* out = z + (((size_t)b * SS + s_) * NKN + nk) * DM;
  ln_row(know + (size_t)nk * DM, proj + row * DM, g, be, out);
}

// =====================================================================
// host-side launch
// =====================================================================
extern "C" void kernel_launch(void* const* d_in, const int* in_sizes, int n_in,
                              void* d_out, int out_size, void* d_ws, size_t ws_size,
                              hipStream_t stream) {
  const float* q_emb   = (const float*)d_in[0];
  const float* s_emb   = (const float*)d_in[1];
  // d_in[2] = lens (int) — unused in eval math
  const float* b1_Wq   = (const float*)d_in[3];
  const float* b1_bq   = (const float*)d_in[4];
  const float* b1_Wv   = (const float*)d_in[5];
  const float* b1_bv   = (const float*)d_in[6];
  const float* b1_Wo   = (const float*)d_in[7];
  const float* b1_bo   = (const float*)d_in[8];
  const float* b1_gam  = (const float*)d_in[9];
  const float* b1_lng  = (const float*)d_in[10];
  const float* b1_lnb  = (const float*)d_in[11];
  const float* b4_Wq   = (const float*)d_in[12];
  const float* b4_bq   = (const float*)d_in[13];
  const float* b4_Wk   = (const float*)d_in[14];
  const float* b4_bk   = (const float*)d_in[15];
  const float* b4_Wv   = (const float*)d_in[16];
  const float* b4_bv   = (const float*)d_in[17];
  const float* b4_Wo   = (const float*)d_in[18];
  const float* b4_bo   = (const float*)d_in[19];
  const float* b4_gam  = (const float*)d_in[20];
  const float* b4_lng  = (const float*)d_in[21];
  const float* b4_lnb  = (const float*)d_in[22];
  const float* know    = (const float*)d_in[23];

  float* out = (float*)d_out;
  const size_t Z_SZ = (size_t)BSB * SS * NKN * DM;   // 2,097,152
  const size_t Q_SZ = (size_t)BSB * HH * SS * SS;    // 2,097,152
  float* z_out  = out;
  float* qs_out = out + Z_SZ;
  float* ks_out = out + Z_SZ + Q_SZ;

  // workspace layout (floats)
  float* ws = (float*)d_ws;
  const size_t R1 = (size_t)BSB * SS * DM;           // 131072
  float* Q1  = ws;                 // BS*S*D
  float* V1  = Q1  + R1;
  float* A1  = V1  + R1;
  float* O1  = A1  + R1;
  float* P   = O1  + R1;
  float* K4  = P   + R1;
  float* V4  = K4  + R1;
  float* QK4 = V4  + R1;           // NK*D = 2048
  float* CB  = QK4 + (size_t)NKN * DM;               // BS*H*NK*S = 131072
  float* A4  = CB  + (size_t)BSB * HH * NKN * SS;    // BS*NK*S*D = 2,097,152
  float* O4  = A4  + (size_t)BSB * NKN * SS * DM;    // 2,097,152

  dim3 w(32);

  // ---------- block 1 ----------
  proj128_kernel<<<dim3(BSB * SS / 16, DM / 16), w, 0, stream>>>(q_emb, b1_Wq, b1_bq, Q1);
  proj128_kernel<<<dim3(BSB * SS / 16, DM / 16), w, 0, stream>>>(s_emb, b1_Wv, b1_bv, V1);
  attn1_kernel<<<dim3(SS / 16, HH, BSB), w, 0, stream>>>(Q1, V1, b1_gam, qs_out, A1);
  proj128_kernel<<<dim3(BSB * SS / 16, DM / 16), w, 0, stream>>>(A1, b1_Wo, b1_bo, O1);
  addln1_kernel<<<dim3(BSB * SS), w, 0, stream>>>(q_emb, O1, b1_lng, b1_lnb, P);

  // ---------- block 4 projections (broadcast structure collapsed) ----------
  proj128_kernel<<<dim3(NKN / 16, DM / 16), w, 0, stream>>>(know, b4_Wq, b4_bq, QK4);
  proj128_kernel<<<dim3(BSB * SS / 16, DM / 16), w, 0, stream>>>(q_emb, b4_Wk, b4_bk, K4);
  proj128_kernel<<<dim3(BSB * SS / 16, DM / 16), w, 0, stream>>>(P, b4_Wv, b4_bv, V4);

  // ---------- block 4 attention ----------
  score4_base_kernel<<<dim3(SS / 16, HH, BSB), w, 0, stream>>>(QK4, K4, CB);
  attn4_kernel<<<dim3(SS / 16, HH, BSB * NKN), w, 0, stream>>>(CB, V4, b4_gam, ks_out, A4);

  // ---------- block 4 output proj + LN -> z ----------
  proj128_kernel<<<dim3(BSB * NKN * SS / 16, DM / 16), w, 0, stream>>>(A4, b4_Wo, b4_bo, O4);
  addln4_kernel<<<dim3(BSB * NKN * SS), w, 0, stream>>>(know, O4, b4_lng, b4_lnb, z_out);
}
